// AttentionalCorrelation_29008209117832
// MI455X (gfx1250) — matrix operand order repaired and checked by reading, not compiled, not run on hardware
//
#include <hip/hip_runtime.h>
#include <hip/hip_bf16.h>

// ---------------------------------------------------------------------------
// AttentionalCorrelation for MI455X (gfx1250, wave32, WMMA bf16 16x16x32).
// B=8, C=128, HID=64, OUT=64, N=H*W=4096.
// Strategy: bf16 WMMA everywhere (f32 accum), flash-attention fusion so the
// [N,N] logits never touch HBM, final projection fused into the same kernel.
// V-tile staging uses the Tensor Data Mover (tensor_load_to_lds) with TDM
// padding reproducing the conflict-free 72-half LDS row stride.
// ---------------------------------------------------------------------------

typedef __attribute__((ext_vector_type(16))) __bf16 v16bf;
typedef __attribute__((ext_vector_type(8)))  float  v8f;
typedef __attribute__((ext_vector_type(4)))  unsigned int v4u;
typedef __attribute__((ext_vector_type(8)))  unsigned int v8u;

union Frag {
  v16bf v;
  uint4 u[2];
  unsigned short s[16];
};

__device__ __forceinline__ unsigned int pack_bf2(float a, float b) {
  union { __bf16 h[2]; unsigned int u; } t;
  t.h[0] = (__bf16)a; t.h[1] = (__bf16)b;
  return t.u;
}
__device__ __forceinline__ unsigned short bf16bits(float x) {
  union { __bf16 h; unsigned short u; } t; t.h = (__bf16)x; return t.u;
}

// ---------------------------------------------------------------------------
// Kernel 1: f32 -> bf16 conversion (V = bf16(F2)). 8 elems/thread, b128 I/O.
// ---------------------------------------------------------------------------
__global__ __launch_bounds__(256) void ac_cvt_kernel(const float* __restrict__ src,
                                                     unsigned short* __restrict__ dst) {
  const int i = (blockIdx.x * 256 + threadIdx.x) * 8;
  const float4* p = (const float4*)(src + i);
  float4 a = p[0], b = p[1];
  uint4 o;
  o.x = pack_bf2(a.x, a.y); o.y = pack_bf2(a.z, a.w);
  o.z = pack_bf2(b.x, b.y); o.w = pack_bf2(b.z, b.w);
  *(uint4*)(dst + i) = o;
}

// ---------------------------------------------------------------------------
// Kernel 2: projections Q = Wq*F1 + bq, K = Wk*F2 + bk  (bf16 out, [d][n]).
// grid = (N/64, B, 2); block = 256 (8 waves). Each block: [64d x 64n] strip.
// ---------------------------------------------------------------------------
__global__ __launch_bounds__(256) void ac_proj_kernel(
    const float* __restrict__ f1, const float* __restrict__ f2,
    const float* __restrict__ wq, const float* __restrict__ bq,
    const float* __restrict__ wk, const float* __restrict__ bk,
    unsigned short* __restrict__ qb, unsigned short* __restrict__ kb) {
  constexpr int N = 4096;
  const int b = blockIdx.y, z = blockIdx.z;
  const float* src  = z ? f2 : f1;
  const float* w    = z ? wk : wq;
  const float* bias = z ? bk : bq;
  unsigned short* dst = z ? kb : qb;

  const int n0  = blockIdx.x * 64;
  const int tid = threadIdx.x;
  const int wave = tid >> 5, lane = tid & 31, l15 = lane & 15, hi = lane >> 4;
  const int dt = wave >> 1;           // d-tile 0..3
  const int ntb = (wave & 1) * 2;     // first of two n-tiles

  __shared__ __align__(16) unsigned short lds_w[64 * 40];  // [d][c'] pad 40
  __shared__ __align__(16) unsigned short lds_f[64 * 40];  // [n][c'] pad 40

  v8f acc[2];
#pragma unroll
  for (int r = 0; r < 8; ++r) {
    float bv = bias[dt * 16 + r + hi * 8];   // bias folded into C operand
    acc[0][r] = bv; acc[1][r] = bv;
  }

  for (int cb = 0; cb < 4; ++cb) {
    const int cbase = cb * 32;
    // stage W tile [64d x 32c] -> lds_w (bf16, row-major)
    {
      int d = tid >> 2, c8 = (tid & 3) * 8;
      const float4* p = (const float4*)(w + d * 128 + cbase + c8);
      float4 a = p[0], bb = p[1];
      uint4 o;
      o.x = pack_bf2(a.x, a.y); o.y = pack_bf2(a.z, a.w);
      o.z = pack_bf2(bb.x, bb.y); o.w = pack_bf2(bb.z, bb.w);
      *(uint4*)&lds_w[d * 40 + c8] = o;
    }
    // stage F tile [32c x 64n] transposed -> lds_f[n][c']
    {
      int c = tid >> 3, n8 = (tid & 7) * 8;
      const float4* p = (const float4*)(src + (size_t)(b * 128 + cbase + c) * N + n0 + n8);
      float4 a = p[0], bb = p[1];
      float vals[8] = {a.x, a.y, a.z, a.w, bb.x, bb.y, bb.z, bb.w};
#pragma unroll
      for (int j = 0; j < 8; ++j) lds_f[(n8 + j) * 40 + c] = bf16bits(vals[j]);
    }
    __syncthreads();

    Frag af;  // A: [16d x 32c], ISA A-layout: K = hi*8+e / 16+hi*8+e
    {
      int drow = dt * 16 + l15;
      af.u[0] = *(const uint4*)&lds_w[drow * 40 + hi * 8];
      af.u[1] = *(const uint4*)&lds_w[drow * 40 + 16 + hi * 8];
    }
#pragma unroll
    for (int t = 0; t < 2; ++t) {
      int nrow = (ntb + t) * 16 + l15;
      Frag bf2_;  // B: [32c x 16n], ISA B-layout: K = hi*8+e / 16+hi*8+e
      bf2_.u[0] = *(const uint4*)&lds_f[nrow * 40 + hi * 8];
      bf2_.u[1] = *(const uint4*)&lds_f[nrow * 40 + 16 + hi * 8];
      acc[t] = __builtin_amdgcn_wmma_f32_16x16x32_bf16(
          false, af.v, false, bf2_.v, (short)0, acc[t], false, false);
    }
    __syncthreads();
  }
  // store bf16 Q/K in [d][n] layout
#pragma unroll
  for (int t = 0; t < 2; ++t) {
    int ncol = n0 + (ntb + t) * 16 + l15;
#pragma unroll
    for (int r = 0; r < 8; ++r) {
      int drow = dt * 16 + r + hi * 8;
      dst[(size_t)(b * 64 + drow) * N + ncol] = bf16bits(acc[t][r]);
    }
  }
}

// ---------------------------------------------------------------------------
// Kernel 3: fused flash attention + output projection.
// grid = (N/128, B); block = 256 (8 waves, 16 query columns each).
// S^T = K^T Q (per-lane-column softmax stats), O = F2 * P^T -> [c,n] layout,
// then out = Wf * O + bf via 16 more WMMAs per wave.
// V tile is DMA'd by the Tensor Data Mover directly into padded LDS.
// LDS unioned: loop phase {K,V,P} 46080 B; epilogue {O slabs, Wf} 52224 B.
// ---------------------------------------------------------------------------
#define AC_SMEM 52224
__global__ __launch_bounds__(256) void ac_flash_kernel(
    const unsigned short* __restrict__ qb, const unsigned short* __restrict__ kb,
    const unsigned short* __restrict__ vb,
    const float* __restrict__ wf, const float* __restrict__ bfv,
    float* __restrict__ outp) {
  constexpr int N = 4096;
  __shared__ __align__(16) char smem[AC_SMEM];
  unsigned short* lds_k  = (unsigned short*)smem;            // [64m][72d]
  unsigned short* lds_v  = (unsigned short*)(smem + 9216);   // [128c][72m]
  unsigned short* lds_p  = (unsigned short*)(smem + 27648);  // 8x [16n][72m]
  unsigned short* lds_o  = (unsigned short*)smem;            // 8x [16n][136c]
  unsigned short* lds_wf = (unsigned short*)(smem + 34816);  // [64o][136c]

  const int b = blockIdx.y;
  const int tid = threadIdx.x;
  const int wave = tid >> 5, lane = tid & 31, l15 = lane & 15, hi = lane >> 4;
  const int nq  = blockIdx.x * 128 + wave * 16;
  const int col = nq + l15;  // this lane's query column

  // --- TDM descriptor for the V tile: 2D tensor [128 x 4096] bf16,
  // tile [tile_dim1=128 rows x tile_dim0=64 cols], dest lds_v with padding:
  // pad_interval=4 (after every 32 DWORDs = one 128B row) pad_amount=3 (16B)
  // -> effective LDS row stride 144B = 72 halves (matches fragment loads).
  v8u tdm_g1;
  tdm_g1[0] = (1u << 16)      // data_size = 2 bytes
            | (1u << 20)      // pad_enable
            | (4u << 22)      // pad_interval: 32 DWORDs
            | (3u << 25);     // pad_amount: 4 DWORDs
  tdm_g1[1] = (4096u & 0xFFFFu) << 16;            // tensor_dim0 lo16 @ [63:48]
  tdm_g1[2] = (4096u >> 16) | ((128u & 0xFFFFu) << 16);  // dim0 hi / dim1 lo
  tdm_g1[3] = (128u >> 16) | (64u << 16);         // dim1 hi / tile_dim0 = 64
  tdm_g1[4] = 128u;                               // tile_dim1 = 128 (dim2 = 0)
  tdm_g1[5] = 4096u;                              // tensor_dim0_stride lo32
  tdm_g1[6] = 0u;                                 // stride hi / dim1_stride lo
  tdm_g1[7] = 0u;
  const unsigned long long vbase =
      (unsigned long long)(size_t)(vb + (size_t)b * 128 * N);
  const unsigned lds_v_off = (unsigned)(size_t)lds_v;  // LDS byte offset

  // Preload Q fragments (B operand, two 32-d chunks), reused for all 64 m-steps
  Frag qf[2];
#pragma unroll
  for (int db = 0; db < 2; ++db)
#pragma unroll
    for (int e = 0; e < 16; ++e) {
      int d = db * 32 + hi * 16 + e;
      qf[db].s[e] = qb[(size_t)(b * 64 + d) * N + col];
    }

  v8f o[8];
#pragma unroll
  for (int ct = 0; ct < 8; ++ct) o[ct] = v8f{};
  float mrun = -INFINITY, lrun = 0.f;

  for (int m0 = 0; m0 < N; m0 += 64) {
    // kick off V-tile DMA (one wave issues; EXEC is ignored by TDM)
    if (wave == 0) {
      unsigned long long ga = vbase + (unsigned long long)m0 * 2u;
      v4u g0;
      g0[0] = 1u;                                   // count=1 (valid, user)
      g0[1] = lds_v_off;                            // lds_addr
      g0[2] = (unsigned)ga;                         // global_addr[31:0]
      g0[3] = (unsigned)((ga >> 32) & 0x01FFFFFFu)  // global_addr[56:32]
            | 0x80000000u;                          // type=2 (image)
      asm volatile("tensor_load_to_lds %0, %1" :: "s"(g0), "s"(tdm_g1) : "memory");
    }
    // stage K tile transposed -> lds_k[m][d] (global coalesced, LDS scattered)
    {
      int d = tid >> 2, mm = (tid & 3) * 16;
      const uint4* p = (const uint4*)(kb + (size_t)(b * 64 + d) * N + m0 + mm);
      Frag t; t.u[0] = p[0]; t.u[1] = p[1];
#pragma unroll
      for (int j = 0; j < 16; ++j) lds_k[(mm + j) * 72 + d] = t.s[j];
    }
    if (wave == 0) __builtin_amdgcn_s_wait_tensorcnt(0);
    __syncthreads();

    // S^T tiles [16m x 16n], 4 m-tiles, K-chunks over d
    v8f sacc[4];
#pragma unroll
    for (int mt = 0; mt < 4; ++mt) {
      sacc[mt] = v8f{};
      int mrow = mt * 16 + l15;
#pragma unroll
      for (int db = 0; db < 2; ++db) {
        Frag af;
        af.u[0] = *(const uint4*)&lds_k[mrow * 72 + db * 32 + hi * 16];
        af.u[1] = *(const uint4*)&lds_k[mrow * 72 + db * 32 + hi * 16 + 8];
        sacc[mt] = __builtin_amdgcn_wmma_f32_16x16x32_bf16(
            false, af.v, false, qf[db].v, (short)0, sacc[mt], false, false);
      }
    }
    // online softmax: per-column stats (column == lane pair {l, l^16})
    float tmax = -INFINITY;
#pragma unroll
    for (int mt = 0; mt < 4; ++mt)
#pragma unroll
      for (int r = 0; r < 8; ++r) tmax = fmaxf(tmax, sacc[mt][r]);
    tmax = fmaxf(tmax, __shfl_xor(tmax, 16));
    float mnew  = fmaxf(mrun, tmax);
    float alpha = __expf(mrun - mnew);
    mrun = mnew;
    float psum = 0.f;
#pragma unroll
    for (int mt = 0; mt < 4; ++mt)
#pragma unroll
      for (int r = 0; r < 8; ++r) {
        float p = __expf(sacc[mt][r] - mnew);
        sacc[mt][r] = p;
        psum += p;
      }
    psum += __shfl_xor(psum, 16);
    lrun = lrun * alpha + psum;
#pragma unroll
    for (int ct = 0; ct < 8; ++ct)
#pragma unroll
      for (int r = 0; r < 8; ++r) o[ct][r] *= alpha;

    // P (C-layout f32) -> wave-private LDS slab [n][m] as bf16
    unsigned short* pw = lds_p + wave * (16 * 72);
#pragma unroll
    for (int mt = 0; mt < 4; ++mt)
#pragma unroll
      for (int j = 0; j < 4; ++j)
        *(unsigned int*)&pw[l15 * 72 + mt * 16 + hi * 8 + 2 * j] =
            pack_bf2(sacc[mt][2 * j], sacc[mt][2 * j + 1]);

    // O += F2_tile * P^T  (D in [c][n] layout)
#pragma unroll
    for (int mc = 0; mc < 2; ++mc) {
      Frag pf;  // B operand [32m x 16n]
      pf.u[0] = *(const uint4*)&pw[l15 * 72 + mc * 32 + hi * 16];
      pf.u[1] = *(const uint4*)&pw[l15 * 72 + mc * 32 + hi * 16 + 8];
#pragma unroll
      for (int ct = 0; ct < 8; ++ct) {
        int c = ct * 16 + l15;
        Frag vf;  // A operand [16c x 32m]
        vf.u[0] = *(const uint4*)&lds_v[c * 72 + mc * 32 + hi * 16];
        vf.u[1] = *(const uint4*)&lds_v[c * 72 + mc * 32 + hi * 16 + 8];
        o[ct] = __builtin_amdgcn_wmma_f32_16x16x32_bf16(
            false, vf.v, false, pf.v, (short)0, o[ct], false, false);
      }
    }
    if (m0 + 64 < N) {  // global_prefetch of next K tile
      __builtin_prefetch(kb + (size_t)(b * 64 + (tid >> 2)) * N + m0 + 64, 0, 1);
    }
    __syncthreads();
  }

  // normalize O by the softmax denominator (per-column, lane-resident)
  float inv = 1.f / lrun;
#pragma unroll
  for (int ct = 0; ct < 8; ++ct)
#pragma unroll
    for (int r = 0; r < 8; ++r) o[ct][r] *= inv;

  // ---- epilogue: out = Wf * O + bf -------------------------------------
  // stage Wf [64o x 128c] -> bf16 LDS
  {
    int orow = tid >> 2, c32 = (tid & 3) * 32;
#pragma unroll
    for (int g = 0; g < 4; ++g) {
      const float4* p = (const float4*)(wf + orow * 128 + c32 + g * 8);
      float4 a = p[0], bb = p[1];
      uint4 u;
      u.x = pack_bf2(a.x, a.y); u.y = pack_bf2(a.z, a.w);
      u.z = pack_bf2(bb.x, bb.y); u.w = pack_bf2(bb.z, bb.w);
      *(uint4*)&lds_wf[orow * 136 + c32 + g * 8] = u;
    }
  }
  // O (C-layout f32) -> wave-private LDS slab [n][c] as bf16
  unsigned short* ow = lds_o + wave * (16 * 136);
#pragma unroll
  for (int ct = 0; ct < 8; ++ct)
#pragma unroll
    for (int j = 0; j < 4; ++j)
      *(unsigned int*)&ow[l15 * 136 + ct * 16 + hi * 8 + 2 * j] =
          pack_bf2(o[ct][2 * j], o[ct][2 * j + 1]);
  __syncthreads();

#pragma unroll
  for (int ot = 0; ot < 4; ++ot) {
    v8f acc;
#pragma unroll
    for (int r = 0; r < 8; ++r) acc[r] = bfv[ot * 16 + r + hi * 8];
#pragma unroll
    for (int ck = 0; ck < 4; ++ck) {
      int cbase = ck * 32;
      int orow = ot * 16 + l15;
      Frag wfr, ofr;
      wfr.u[0] = *(const uint4*)&lds_wf[orow * 136 + cbase + hi * 16];
      wfr.u[1] = *(const uint4*)&lds_wf[orow * 136 + cbase + hi * 16 + 8];
      ofr.u[0] = *(const uint4*)&ow[l15 * 136 + cbase + hi * 16];
      ofr.u[1] = *(const uint4*)&ow[l15 * 136 + cbase + hi * 16 + 8];
      acc = __builtin_amdgcn_wmma_f32_16x16x32_bf16(
          false, wfr.v, false, ofr.v, (short)0, acc, false, false);
    }
#pragma unroll
    for (int r = 0; r < 8; ++r)
      outp[(size_t)(b * 64 + ot * 16 + r + hi * 8) * N + col] = acc[r];
  }
}

// ---------------------------------------------------------------------------
extern "C" void kernel_launch(void* const* d_in, const int* in_sizes, int n_in,
                              void* d_out, int out_size, void* d_ws, size_t ws_size,
                              hipStream_t stream) {
  const float* f1 = (const float*)d_in[0];
  const float* f2 = (const float*)d_in[1];
  const float* wq = (const float*)d_in[2];
  const float* bq = (const float*)d_in[3];
  const float* wk = (const float*)d_in[4];
  const float* bk = (const float*)d_in[5];
  const float* wf = (const float*)d_in[6];
  const float* bf = (const float*)d_in[7];
  float* out = (float*)d_out;

  // workspace: Q bf16 (4MB) | K bf16 (4MB) | V bf16 (8MB)  => 16MB
  unsigned short* qb = (unsigned short*)d_ws;
  unsigned short* kb = qb + (size_t)8 * 64 * 4096;
  unsigned short* vb = kb + (size_t)8 * 64 * 4096;

  // V = bf16(F2): 8*128*4096 = 4194304 elems, 8/thread
  ac_cvt_kernel<<<2048, 256, 0, stream>>>(f2, vb);
  // Q, K projections (z=0 -> Q from F1/Wq/bq, z=1 -> K from F2/Wk/bk)
  ac_proj_kernel<<<dim3(64, 8, 2), 256, 0, stream>>>(f1, f2, wq, bq, wk, bk, qb, kb);
  // fused flash attention + output projection
  ac_flash_kernel<<<dim3(32, 8), 256, 0, stream>>>(qb, kb, vb, wf, bf, out);
}
